// BiLSTM_CRF_47674136985653
// MI455X (gfx1250) — matrix operand order repaired and checked by reading, not compiled, not run on hardware
//
#include <hip/hip_runtime.h>
#include <math.h>

#define VOCAB 30000
#define TAG   17
#define EMB   100
#define HID   128
#define BB    256
#define LL    512
#define BL    (BB * LL)   // 131072
#define KCAT  256         // padded E (128) + HID (128)
#define GATE  512         // 4*HID

typedef _Float16 v8h  __attribute__((ext_vector_type(8)));
typedef _Float16 v16h __attribute__((ext_vector_type(16)));
typedef float    v8f  __attribute__((ext_vector_type(8)));

// ---------------------------------------------------------------------------
// WMMA fragment loaders (layouts per CDNA5 ISA 7.12.2, 16-bit, 16x16x32)
// ---------------------------------------------------------------------------
__device__ __forceinline__ v16h load_frag_a(const _Float16* rowp, int laneHalf) {
    const _Float16* q = rowp + laneHalf * 8;
    v8h lo = *(const v8h*)(q);
    v8h hi = *(const v8h*)(q + 16);
    return __builtin_shufflevector(lo, hi, 0,1,2,3,4,5,6,7,8,9,10,11,12,13,14,15);
}

__device__ __forceinline__ v16h load_frag_b(const _Float16* rowp, int laneHalf) {
    const _Float16* q = rowp + laneHalf * 8;
    v8h lo = *(const v8h*)(q);
    v8h hi = *(const v8h*)(q + 16);
    return __builtin_shufflevector(lo, hi, 0,1,2,3,4,5,6,7,8,9,10,11,12,13,14,15);
}

// Fast saturating sigmoid/tanh: v_exp_f32 + v_rcp_f32 (no div_fixup chains).
__device__ __forceinline__ float fast_sigmoid(float x) {
    return __builtin_amdgcn_rcpf(1.0f + __expf(-x));
}
__device__ __forceinline__ float fast_tanh(float x) {
    return 1.0f - 2.0f * __builtin_amdgcn_rcpf(__expf(2.0f * x) + 1.0f);
}

// ---------------------------------------------------------------------------
// Kernel 1: weight prep -> fp16 (concat + pad layouts for WMMA B operand)
// ---------------------------------------------------------------------------
__global__ __launch_bounds__(256) void prep_weights_kernel(
    const float* __restrict__ W_ih_f, const float* __restrict__ W_hh_f,
    const float* __restrict__ W_ih_b, const float* __restrict__ W_hh_b,
    const float* __restrict__ fc_W,
    _Float16* __restrict__ Wcat_f, _Float16* __restrict__ Wcat_b,
    _Float16* __restrict__ fcW)
{
    int g = blockIdx.x * 256 + threadIdx.x;
    if (g < GATE * KCAT) {
        int row = g >> 8, col = g & 255;
        float vf, vb;
        if (col < 128) {
            vf = (col < EMB) ? W_ih_f[row * EMB + col] : 0.0f;
            vb = (col < EMB) ? W_ih_b[row * EMB + col] : 0.0f;
        } else {
            vf = W_hh_f[row * HID + (col - 128)];
            vb = W_hh_b[row * HID + (col - 128)];
        }
        Wcat_f[g] = (_Float16)vf;
        Wcat_b[g] = (_Float16)vb;
    }
    if (g < 32 * KCAT) {
        int row = g >> 8, col = g & 255;
        fcW[g] = (_Float16)((row < TAG) ? fc_W[row * (2 * HID) + col] : 0.0f);
    }
}

// ---------------------------------------------------------------------------
// Kernel 2: embedding gather -> fp16 X[B*L][128] (E padded to 128)
// ---------------------------------------------------------------------------
__global__ __launch_bounds__(256) void embed_kernel(
    const int* __restrict__ ids, const float* __restrict__ emb,
    _Float16* __restrict__ X)
{
    int g = blockIdx.x * 256 + threadIdx.x;
    int tok = g >> 7, e = g & 127;
    int id = ids[tok];
    float v = (e < EMB) ? emb[id * EMB + e] : 0.0f;
    X[g] = (_Float16)v;
}

// ---------------------------------------------------------------------------
// Kernel 3: recurrent BiLSTM. grid = (16 batch tiles, 2 directions),
// 256 threads = 8 waves. Wave w owns hidden columns [w*16,w*16+16) of all
// four gates -> gate math + c state entirely in registers; one barrier/step.
//   - 8 independent WMMA accumulator chains (accX bias-init / accH zero-init)
//     => dependency distance 8, no f16 WMMA RAW hazard NOPs
//   - time loop unrolled x2 with ping-pong x-fragment sets => prefetched
//     global loads are first consumed a full step after issue (no rotation
//     copies, no early s_wait_loadcnt)
//   - hcat global stores deferred one step so barriers never wait on HBM
// ---------------------------------------------------------------------------
__global__ __launch_bounds__(256) void lstm_kernel(
    const _Float16* __restrict__ X,       // [BL][128]
    const _Float16* __restrict__ Wcat_f,  // [512][256]
    const _Float16* __restrict__ Wcat_b,  // [512][256]
    const float* __restrict__ b_f, const float* __restrict__ b_b,
    _Float16* __restrict__ hcat)          // [BL][256]
{
    __shared__ __align__(16) _Float16 h_s[2][16][HID];   // ping-pong, 8 KB

    const int dir  = blockIdx.y;
    const int b0   = blockIdx.x * 16;
    const _Float16* W   = dir ? Wcat_b : Wcat_f;
    const float*    bia = dir ? b_b    : b_f;

    const int tid  = threadIdx.x;
    const int wave = tid >> 5;
    const int lane = tid & 31;
    const int laneHalf = lane >> 4;
    const int r16      = lane & 15;
    const int colh     = wave * 16 + r16;     // hidden index j owned by lane

    for (int i = tid; i < 16 * HID; i += 256)
        ((_Float16*)h_s[0])[i] = (_Float16)0.0f;

    // ---- weight fragments + bias splat, pinned for the whole loop ---------
    v16h wfrag[4][8];
    float bv[4];
    #pragma unroll
    for (int gi = 0; gi < 4; ++gi) {
        const int col = gi * 128 + colh;
        const _Float16* wrow = W + col * KCAT;
        #pragma unroll
        for (int ks = 0; ks < 8; ++ks)
            wfrag[gi][ks] = load_frag_b(wrow + ks * 32, laneHalf);
        bv[gi] = bia[col];
    }

    float creg[8];
    #pragma unroll
    for (int v = 0; v < 8; ++v) creg[v] = 0.0f;

    const _Float16* xbase = X + (size_t)(b0 + r16) * LL * 128;
    const int t0 = dir ? (LL - 1) : 0;

    v16h xA[4], xB[4];
    #pragma unroll
    for (int ks = 0; ks < 4; ++ks)
        xA[ks] = load_frag_a(xbase + t0 * 128 + ks * 32, laneHalf);

    float hprev[8];
    int   tprev = 0;
    int   p = 0;
    __syncthreads();

    auto step_body = [&](int step, v16h (&xcur)[4], v16h (&xnxt)[4]) {
        const int t = dir ? (LL - 1 - step) : step;
        const int stepn = (step + 1 < LL) ? (step + 1) : step;
        const int tn = dir ? (LL - 1 - stepn) : stepn;

        // deferred global store of last step's h (wait hidden behind GEMM)
        if (step > 0) {
            #pragma unroll
            for (int v = 0; v < 8; ++v) {
                const int row = v + 8 * laneHalf;
                hcat[((size_t)(b0 + row) * LL + tprev) * (2 * HID) + dir * HID + colh] =
                    (_Float16)hprev[v];
            }
        }

        // prefetch step+1's x fragments into the OTHER register set;
        // first consumed one full step from now.
        #pragma unroll
        for (int ks = 0; ks < 4; ++ks)
            xnxt[ks] = load_frag_a(xbase + tn * 128 + ks * 32, laneHalf);

        // h_{t-1} fragments from LDS buffer p
        v16h hf[4];
        #pragma unroll
        for (int ks = 0; ks < 4; ++ks)
            hf[ks] = load_frag_a(&h_s[p][r16][ks * 32], laneHalf);

        // ---- 8 independent WMMA chains: accX (bias-init) + accH (0-init) --
        v8f accX[4], accH[4];
        #pragma unroll
        for (int gi = 0; gi < 4; ++gi) {
            accX[gi] = (v8f){bv[gi], bv[gi], bv[gi], bv[gi],
                             bv[gi], bv[gi], bv[gi], bv[gi]};
            accH[gi] = (v8f){0.f, 0.f, 0.f, 0.f, 0.f, 0.f, 0.f, 0.f};
        }
        #pragma unroll
        for (int ks = 0; ks < 4; ++ks) {
            #pragma unroll
            for (int gi = 0; gi < 4; ++gi)           // K 0..127 : x_t
                accX[gi] = __builtin_amdgcn_wmma_f32_16x16x32_f16(
                    false, xcur[ks], false, wfrag[gi][ks], (short)0, accX[gi], false, false);
            #pragma unroll
            for (int gi = 0; gi < 4; ++gi)           // K 128..255 : h_{t-1}
                accH[gi] = __builtin_amdgcn_wmma_f32_16x16x32_f16(
                    false, hf[ks], false, wfrag[gi][4 + ks], (short)0, accH[gi], false, false);
        }

        // ---- gate math entirely in registers ------------------------------
        v8f gI = accX[0] + accH[0];
        v8f gF = accX[1] + accH[1];
        v8f gG = accX[2] + accH[2];
        v8f gO = accX[3] + accH[3];
        #pragma unroll
        for (int v = 0; v < 8; ++v) {
            float cn = fast_sigmoid(gF[v]) * creg[v]
                     + fast_sigmoid(gI[v]) * fast_tanh(gG[v]);
            float hn = fast_sigmoid(gO[v]) * fast_tanh(cn);
            creg[v]  = cn;
            hprev[v] = hn;
            h_s[p ^ 1][v + 8 * laneHalf][colh] = (_Float16)hn;
        }
        tprev = t;
        p ^= 1;
        __syncthreads();
    };

    for (int step = 0; step < LL; step += 2) {
        step_body(step,     xA, xB);
        step_body(step + 1, xB, xA);
    }

    // drain the final h store
    #pragma unroll
    for (int v = 0; v < 8; ++v) {
        const int row = v + 8 * laneHalf;
        hcat[((size_t)(b0 + row) * LL + tprev) * (2 * HID) + dir * HID + colh] =
            (_Float16)hprev[v];
    }
}

// ---------------------------------------------------------------------------
// Kernel 4: FC emissions. em[BL][32] = hcat[BL][256] @ fcW^T + fc_b (cols<17)
// ---------------------------------------------------------------------------
__global__ __launch_bounds__(256) void fc_kernel(
    const _Float16* __restrict__ hcat, const _Float16* __restrict__ fcW,
    const float* __restrict__ fc_b, float* __restrict__ em)
{
    const int tid  = threadIdx.x;
    const int wave = tid >> 5;
    const int lane = tid & 31;
    const int laneHalf = lane >> 4;
    const int r16      = lane & 15;
    const int m0 = blockIdx.x * 128 + wave * 16;
    const _Float16* arow = hcat + (m0 + r16) * KCAT;

    #pragma unroll
    for (int nt = 0; nt < 2; ++nt) {
        const _Float16* wrow = fcW + (nt * 16 + r16) * KCAT;
        v8f acc = {0.f, 0.f, 0.f, 0.f, 0.f, 0.f, 0.f, 0.f};
        #pragma unroll
        for (int ks = 0; ks < 8; ++ks) {
            v16h a  = load_frag_a(arow + ks * 32, laneHalf);
            v16h bm = load_frag_b(wrow + ks * 32, laneHalf);
            acc = __builtin_amdgcn_wmma_f32_16x16x32_f16(
                    false, a, false, bm, (short)0, acc, false, false);
        }
        const int col = nt * 16 + r16;
        const float bvv = (col < TAG) ? fc_b[col] : 0.0f;
        #pragma unroll
        for (int v = 0; v < 8; ++v)
            em[(m0 + v + 8 * laneHalf) * 32 + col] = acc[v] + bvv;
    }
}

// ---------------------------------------------------------------------------
// Kernel 5: Viterbi. One 32-lane wave per batch row; lane = tag (17 active).
// Backpointer history kept entirely in LDS (16 KB).
// ---------------------------------------------------------------------------
__global__ __launch_bounds__(32) void viterbi_kernel(
    const float* __restrict__ em, const int* __restrict__ ids,
    const float* __restrict__ trans, const float* __restrict__ start_t,
    const float* __restrict__ end_t, int* __restrict__ out)
{
    const int b = blockIdx.x, j = threadIdx.x;
    __shared__ float score[32];
    __shared__ float tr_s[TAG * TAG];
    __shared__ unsigned char hist_s[LL][32];

    for (int i = j; i < TAG * TAG; i += 32) tr_s[i] = trans[i];
    const float* emb_ = em + b * LL * 32;
    if (j < TAG) score[j] = start_t[j] + emb_[j];
    __syncthreads();

    for (int t = 1; t < LL; ++t) {
        const bool m = ids[b * LL + t] != 0;
        float best = -3.4e38f;
        int   bi   = j;
        if (j < TAG) {
            #pragma unroll
            for (int i = 0; i < TAG; ++i) {
                float v = score[i] + tr_s[i * TAG + j];
                if (v > best) { best = v; bi = i; }
            }
            best += emb_[t * 32 + j];
        }
        __syncthreads();
        if (j < TAG) {
            if (m) { score[j] = best; hist_s[t][j] = (unsigned char)bi; }
            else   {                  hist_s[t][j] = (unsigned char)j;  }
        }
        __syncthreads();
    }

    if (j < TAG) score[j] += end_t[j];
    __syncthreads();

    if (j == 0) {
        int tag = 0; float bvv = score[0];
        for (int i = 1; i < TAG; ++i)
            if (score[i] > bvv) { bvv = score[i]; tag = i; }
        out[b * LL + (LL - 1)] = (ids[b * LL + (LL - 1)] != 0) ? tag : 0;
        for (int t = LL - 2; t >= 0; --t) {
            tag = hist_s[t + 1][tag];
            out[b * LL + t] = (ids[b * LL + t] != 0) ? tag : 0;
        }
    }
}

// ---------------------------------------------------------------------------
extern "C" void kernel_launch(void* const* d_in, const int* in_sizes, int n_in,
                              void* d_out, int out_size, void* d_ws, size_t ws_size,
                              hipStream_t stream)
{
    const int*   input_ids = (const int*)  d_in[0];
    const float* emb       = (const float*)d_in[1];
    const float* W_ih_f    = (const float*)d_in[2];
    const float* W_hh_f    = (const float*)d_in[3];
    const float* b_f       = (const float*)d_in[4];
    const float* W_ih_b    = (const float*)d_in[5];
    const float* W_hh_b    = (const float*)d_in[6];
    const float* b_b       = (const float*)d_in[7];
    const float* fc_W      = (const float*)d_in[8];
    const float* fc_b      = (const float*)d_in[9];
    const float* trans     = (const float*)d_in[10];
    const float* start_t   = (const float*)d_in[11];
    const float* end_t     = (const float*)d_in[12];

    char* ws = (char*)d_ws;
    size_t off = 0;
    _Float16* X      = (_Float16*)(ws + off); off += (size_t)BL * 128 * 2;   // 32 MB
    _Float16* Wcat_f = (_Float16*)(ws + off); off += (size_t)GATE * KCAT * 2;
    _Float16* Wcat_b = (_Float16*)(ws + off); off += (size_t)GATE * KCAT * 2;
    _Float16* fcW    = (_Float16*)(ws + off); off += (size_t)32 * KCAT * 2;
    _Float16* hcat   = (_Float16*)(ws + off); off += (size_t)BL * 256 * 2;   // 64 MB
    float*    emis   = (float*)   (ws + off); off += (size_t)BL * 32 * 4;    // 16 MB
    int*      out    = (int*)d_out;

    prep_weights_kernel<<<(GATE * KCAT + 255) / 256, 256, 0, stream>>>(
        W_ih_f, W_hh_f, W_ih_b, W_hh_b, fc_W, Wcat_f, Wcat_b, fcW);

    embed_kernel<<<(BL * 128) / 256, 256, 0, stream>>>(input_ids, emb, X);

    lstm_kernel<<<dim3(BB / 16, 2), 256, 0, stream>>>(
        X, Wcat_f, Wcat_b, b_f, b_b, hcat);

    fc_kernel<<<BL / 128, 256, 0, stream>>>(hcat, fcW, fc_b, emis);

    viterbi_kernel<<<BB, 32, 0, stream>>>(
        emis, input_ids, trans, start_t, end_t, out);
}
